// PromptTransformer_48593259987312
// MI455X (gfx1250) — compile-verified, matrix operand order checked by reading, not run on hardware
//
#include <hip/hip_runtime.h>
#include <hip/hip_bf16.h>
#include <math.h>

// ---------------- problem constants ----------------
constexpr int VOCAB = 32000;
constexpr int DMODEL = 1024;
constexpr int NHEAD = 16;
constexpr int FFDIM = 4096;
constexpr int NLAYER = 4;
constexpr int PLEN = 32;
constexpr int BATCH = 4;
constexpr int TLEN = 1024;
constexpr int TTOT = TLEN + PLEN;    // 1056
constexpr int HDIM = DMODEL / NHEAD; // 64

typedef __attribute__((ext_vector_type(16))) __bf16 v16bf;
typedef __attribute__((ext_vector_type(8)))  float  v8f;
typedef __attribute__((ext_vector_type(4)))  unsigned int v4u;

// ---------------- embedding + prompt concat ----------------
__global__ void __launch_bounds__(256) embed_concat_kernel(
    const int* __restrict__ ids, const float* __restrict__ embed,
    const float* __restrict__ prompt, float* __restrict__ x) {
  size_t idx = (size_t)blockIdx.x * 256 + threadIdx.x;
  size_t total = (size_t)BATCH * TTOT * DMODEL;
  if (idx >= total) return;
  int d = (int)(idx % DMODEL);
  size_t bt = idx / DMODEL;
  int t = (int)(bt % TTOT);
  int b = (int)(bt / TTOT);
  float val;
  if (t < PLEN) {
    val = prompt[(size_t)t * DMODEL + d];
  } else {
    int tok = ids[(size_t)b * TLEN + (t - PLEN)];
    val = embed[(size_t)tok * DMODEL + d];
  }
  x[idx] = val;
}

// ---------------- weight transpose + f32->bf16 ----------------
// out[n*K + k] = (bf16) in[k*N + n]; blockIdx.z selects matrix in a stack.
__global__ void __launch_bounds__(256) transpose_bf16_kernel(
    const float* __restrict__ in, __bf16* __restrict__ out, int K, int N) {
  __shared__ float tile[32][33];
  int k0 = blockIdx.x * 32, n0 = blockIdx.y * 32;
  const float* src = in + (size_t)blockIdx.z * K * N;
  __bf16* dst = out + (size_t)blockIdx.z * K * N;
  int tx = threadIdx.x & 31, ty = threadIdx.x >> 5; // 32 x 8
  for (int r = ty; r < 32; r += 8)
    tile[r][tx] = src[(size_t)(k0 + r) * N + (n0 + tx)];
  __syncthreads();
  for (int r = ty; r < 32; r += 8)
    dst[(size_t)(n0 + r) * K + (k0 + tx)] = (__bf16)tile[tx][r];
}

// ---------------- layernorm (one block per row), bf16 output ----------------
__global__ void __launch_bounds__(256) layernorm_kernel(
    const float* __restrict__ x, const float* __restrict__ g,
    const float* __restrict__ bb, __bf16* __restrict__ y) {
  __shared__ float red[256];
  int tid = threadIdx.x;
  const float* xr = x + (size_t)blockIdx.x * DMODEL;
  __bf16* yr = y + (size_t)blockIdx.x * DMODEL;

  float s = 0.f;
  for (int i = tid; i < DMODEL; i += 256) s += xr[i];
  red[tid] = s; __syncthreads();
  for (int off = 128; off > 0; off >>= 1) {
    if (tid < off) red[tid] += red[tid + off];
    __syncthreads();
  }
  float mean = red[0] / DMODEL;
  __syncthreads();

  float vsum = 0.f;
  for (int i = tid; i < DMODEL; i += 256) {
    float d = xr[i] - mean; vsum += d * d;
  }
  red[tid] = vsum; __syncthreads();
  for (int off = 128; off > 0; off >>= 1) {
    if (tid < off) red[tid] += red[tid + off];
    __syncthreads();
  }
  float rstd = rsqrtf(red[0] / DMODEL + 1e-5f);

  for (int i = tid; i < DMODEL; i += 256)
    yr[i] = (__bf16)((xr[i] - mean) * rstd * g[i] + bb[i]);
}

// ---------------- WMMA bf16 GEMM with async-LDS staging ----------------
// C[M,N] = act(A[M,K] @ Bt[N,K]^T + bias) (+resid).
// A, Bt are bf16, K-major rows. 256 threads (8 waves), 128x128 tile, BK=32,
// double-buffered LDS filled with global_load_async_to_lds_b64 (ASYNCcnt).
// Scheduling groups keep DS reads 2 fragments ahead of each WMMA so LDS
// latency hides under the matrix pipe instead of full s_wait_dscnt drains.

#define BM 128
#define BN 128
#define BK 32
#define LDSROW 20      // dwords per 32-bf16 row: 16 data + 4 pad (80B, 16B aligned)

#define GEMM_RELU 1
#define GEMM_OUT_BF16 2

__device__ __forceinline__ void stage_tile_async(
    const __bf16* __restrict__ g, int ld, int row0, int k0,
    unsigned int* lds, int tid) {
  // 128 rows x 32 bf16 (64B) = 1024 chunks of 8B; 4 chunks per thread.
#pragma unroll
  for (int it = 0; it < 4; ++it) {
    int idx = tid + it * 256;
    int r = idx >> 3, c = idx & 7;
    const __bf16* src = g + (size_t)(row0 + r) * ld + k0 + c * 4;
    unsigned int dst = (unsigned int)(uintptr_t)(lds + r * LDSROW + c * 2);
    asm volatile("global_load_async_to_lds_b64 %0, %1, off"
                 :: "v"(dst), "v"(src) : "memory");
  }
}

__device__ __forceinline__ v16bf load_frag(const unsigned int* row, int g) {
  // ISA 7.12.2 16-bit A/B layout: lane half g holds K dword-pairs
  // [g*4 .. g*4+3] and [8+g*4 .. 8+g*4+3] of its 16-dword row.
  union { v16bf bf; v4u u[2]; } f;
  f.u[0] = *(const v4u*)(row + g * 4);
  f.u[1] = *(const v4u*)(row + 8 + g * 4);
  return f.bf;
}

__global__ void __launch_bounds__(256) gemm_bf16_wmma_kernel(
    const __bf16* __restrict__ A, const __bf16* __restrict__ Bt,
    void* __restrict__ Cout, const float* __restrict__ bias,
    const float* __restrict__ resid, int M, int N, int K, int flags) {
  __shared__ unsigned int As[2][BM * LDSROW];
  __shared__ unsigned int Bs[2][BN * LDSROW];

  int tid  = threadIdx.x;
  int lane = tid & 31;
  int wave = tid >> 5;
  int g    = lane >> 4;
  int lm   = lane & 15;
  int m0   = blockIdx.y * BM;
  int n0   = blockIdx.x * BN;

  v8f acc[8] = {};

  // prime buffer 0
  stage_tile_async(A,  K, m0, 0, As[0], tid);
  stage_tile_async(Bt, K, n0, 0, Bs[0], tid);
  asm volatile("s_wait_asynccnt 0x0" ::: "memory");
  __syncthreads();

  int buf = 0;
  for (int k0 = 0; k0 < K; k0 += BK) {
    int nk = k0 + BK;
    if (nk < K) {
      stage_tile_async(A,  K, m0, nk, As[buf ^ 1], tid);
      stage_tile_async(Bt, K, n0, nk, Bs[buf ^ 1], tid);
    }

    // issue all fragment loads, then all WMMAs; sched groups below interleave
    v16bf afrag = load_frag(&As[buf][(wave * 16 + lm) * LDSROW], g);
    v16bf bfrag[8];
#pragma unroll
    for (int t = 0; t < 8; ++t)
      bfrag[t] = load_frag(&Bs[buf][(t * 16 + lm) * LDSROW], g);

#pragma unroll
    for (int t = 0; t < 8; ++t)
      acc[t] = __builtin_amdgcn_wmma_f32_16x16x32_bf16(
          false, afrag, false, bfrag[t], (short)0, acc[t], false, false);

#if __has_builtin(__builtin_amdgcn_sched_group_barrier)
    // Desired order: 6 DS reads (A-frag + 2 B-frags), then (WMMA, 2 DS)x7,
    // then the final WMMA -> DS loads stay 2 fragments ahead of each WMMA.
    __builtin_amdgcn_sched_group_barrier(0x100, 6, 0); // DS read x6
#pragma unroll
    for (int t = 0; t < 7; ++t) {
      __builtin_amdgcn_sched_group_barrier(0x008, 1, 0); // MFMA/WMMA x1
      __builtin_amdgcn_sched_group_barrier(0x100, 2, 0); // DS read x2
    }
    __builtin_amdgcn_sched_group_barrier(0x008, 1, 0);   // final WMMA
#endif

    asm volatile("s_wait_asynccnt 0x0" ::: "memory");
    __syncthreads();
    buf ^= 1;
  }

  // epilogue: C/D layout: lane(0-15)->N, VGPR r -> M=r (upper lane half: M=r+8)
  float* cf = (float*)Cout;
  __bf16* cb = (__bf16*)Cout;
  int mw = m0 + wave * 16;
#pragma unroll
  for (int t = 0; t < 8; ++t) {
    int col = n0 + t * 16 + lm;
    float bv = bias ? bias[col] : 0.f;
#pragma unroll
    for (int r = 0; r < 8; ++r) {
      int row = mw + (g ? r + 8 : r);
      float val = acc[t][r] + bv;
      if (flags & GEMM_RELU) val = val > 0.f ? val : 0.f;
      size_t idx = (size_t)row * N + col;
      if (flags & GEMM_OUT_BF16) {
        cb[idx] = (__bf16)val;
      } else {
        if (resid) val += resid[idx];
        cf[idx] = val;
      }
    }
  }
}

// ---------------- attention (one block per (b,h,q-row)), bf16 out ----------------
__global__ void __launch_bounds__(128) attention_kernel(
    const float* __restrict__ Q, const float* __restrict__ Kk,
    const float* __restrict__ Vv, const float* __restrict__ mask,
    __bf16* __restrict__ O) {
  __shared__ float qs[HDIM];
  __shared__ float sc[TTOT];
  __shared__ float red[128];

  int tq = blockIdx.x, h = blockIdx.y, b = blockIdx.z;
  int tid = threadIdx.x;
  const float scale = 0.125f; // 1/sqrt(64)

  if (tid < HDIM)
    qs[tid] = Q[((size_t)b * TTOT + tq) * DMODEL + h * HDIM + tid];
  __syncthreads();

  for (int k = tid; k < TTOT; k += 128) {
    const float* kr = Kk + ((size_t)b * TTOT + k) * DMODEL + h * HDIM;
    float s = 0.f;
#pragma unroll 8
    for (int d = 0; d < HDIM; ++d) s += qs[d] * kr[d];
    float mval = (k < PLEN) ? 1.f : mask[(size_t)b * TLEN + (k - PLEN)];
    sc[k] = (mval == 0.f) ? -INFINITY : s * scale;
  }
  __syncthreads();

  float lm = -INFINITY;
  for (int k = tid; k < TTOT; k += 128) lm = fmaxf(lm, sc[k]);
  red[tid] = lm; __syncthreads();
  for (int off = 64; off > 0; off >>= 1) {
    if (tid < off) red[tid] = fmaxf(red[tid], red[tid + off]);
    __syncthreads();
  }
  float mx = red[0];
  __syncthreads();

  float ls = 0.f;
  for (int k = tid; k < TTOT; k += 128) {
    float p = __expf(sc[k] - mx);
    sc[k] = p; ls += p;
  }
  __syncthreads();
  red[tid] = ls; __syncthreads();
  for (int off = 64; off > 0; off >>= 1) {
    if (tid < off) red[tid] += red[tid + off];
    __syncthreads();
  }
  float inv_sum = 1.f / red[0];
  __syncthreads();

  if (tid < HDIM) {
    float acc = 0.f;
    const float* vb = Vv + (size_t)b * TTOT * DMODEL + h * HDIM + tid;
    for (int k = 0; k < TTOT; ++k)
      acc += sc[k] * vb[(size_t)k * DMODEL];
    O[((size_t)b * TTOT + tq) * DMODEL + h * HDIM + tid] =
        (__bf16)(acc * inv_sum);
  }
}

// ---------------- loss ----------------
__global__ void __launch_bounds__(256) loss_row_kernel(
    const float* __restrict__ logits, const int* __restrict__ labels,
    float* __restrict__ rowloss) {
  __shared__ float red[256];
  int row = blockIdx.x; // 0..B*T-1
  int b = row / TLEN, t = row % TLEN;
  int tid = threadIdx.x;
  const float* lr = logits + ((size_t)b * TTOT + PLEN + t) * VOCAB;

  float m = -INFINITY;
  for (int i = tid; i < VOCAB; i += 256) m = fmaxf(m, lr[i]);
  red[tid] = m; __syncthreads();
  for (int off = 128; off > 0; off >>= 1) {
    if (tid < off) red[tid] = fmaxf(red[tid], red[tid + off]);
    __syncthreads();
  }
  float mx = red[0];
  __syncthreads();

  float s = 0.f;
  for (int i = tid; i < VOCAB; i += 256) s += __expf(lr[i] - mx);
  red[tid] = s; __syncthreads();
  for (int off = 128; off > 0; off >>= 1) {
    if (tid < off) red[tid] += red[tid + off];
    __syncthreads();
  }
  if (tid == 0) {
    int lab = labels[row];
    rowloss[row] = -(lr[lab] - mx - __logf(red[0]));
  }
}

__global__ void __launch_bounds__(256) loss_final_kernel(
    const float* __restrict__ rowloss, float* __restrict__ out) {
  __shared__ float red[256];
  int tid = threadIdx.x;
  float s = 0.f;
  for (int i = tid; i < BATCH * TLEN; i += 256) s += rowloss[i];
  red[tid] = s; __syncthreads();
  for (int off = 128; off > 0; off >>= 1) {
    if (tid < off) red[tid] += red[tid + off];
    __syncthreads();
  }
  if (tid == 0) out[0] = red[0] / (float)(BATCH * TLEN);
}

// ---------------- host orchestration ----------------
static inline void launch_gemm(const __bf16* A, const __bf16* Bt, void* C,
                               const float* bias, const float* resid,
                               int M, int N, int K, int flags,
                               hipStream_t stream) {
  dim3 grid(N / BN, M / BM);
  gemm_bf16_wmma_kernel<<<grid, 256, 0, stream>>>(A, Bt, C, bias, resid,
                                                  M, N, K, flags);
}

extern "C" void kernel_launch(void* const* d_in, const int* in_sizes, int n_in,
                              void* d_out, int out_size, void* d_ws, size_t ws_size,
                              hipStream_t stream) {
  const int*   input_ids = (const int*)  d_in[0];
  const float* attn_mask = (const float*)d_in[1];
  const int*   labels    = (const int*)  d_in[2];
  const float* embed     = (const float*)d_in[3];
  const float* prompt    = (const float*)d_in[4];
  const float* Wq        = (const float*)d_in[5];
  const float* Wk        = (const float*)d_in[6];
  const float* Wv        = (const float*)d_in[7];
  const float* Wo        = (const float*)d_in[8];
  const float* bq        = (const float*)d_in[9];
  const float* bk        = (const float*)d_in[10];
  const float* bv        = (const float*)d_in[11];
  const float* bo        = (const float*)d_in[12];
  const float* ln1_g     = (const float*)d_in[13];
  const float* ln1_b     = (const float*)d_in[14];
  const float* ln2_g     = (const float*)d_in[15];
  const float* ln2_b     = (const float*)d_in[16];
  const float* W1        = (const float*)d_in[17];
  const float* b1        = (const float*)d_in[18];
  const float* W2        = (const float*)d_in[19];
  const float* b2        = (const float*)d_in[20];
  const float* lnf_g     = (const float*)d_in[21];
  const float* lnf_b     = (const float*)d_in[22];
  const float* head_W    = (const float*)d_in[23];

  const int M = BATCH * TTOT;          // 4224 rows
  const size_t X = (size_t)M * DMODEL; // activation elements

  // ---- workspace carve-up ----
  float* x       = (float*)d_ws;
  float* q       = x + X;
  float* k       = q + X;
  float* v       = k + X;
  float* rowloss = v + X;              // 4096 floats
  __bf16* h    = (__bf16*)(rowloss + 4096);
  __bf16* o    = h + X;
  __bf16* ffh  = o + X;                // M x FF = 4X elems
  const size_t SZ_DD = (size_t)NLAYER * DMODEL * DMODEL;
  const size_t SZ_DF = (size_t)NLAYER * DMODEL * FFDIM;
  __bf16* WqT  = ffh + 4 * X;
  __bf16* WkT  = WqT + SZ_DD;
  __bf16* WvT  = WkT + SZ_DD;
  __bf16* WoT  = WvT + SZ_DD;
  __bf16* W1T  = WoT + SZ_DD;
  __bf16* W2T  = W1T + SZ_DF;
  __bf16* headT = W2T + SZ_DF;         // VOCAB x DMODEL

  float* logits = (float*)d_out;
  float* lossp  = logits + (size_t)BATCH * TTOT * VOCAB;

  // ---- 0) weight prep: transpose + convert to bf16 [N,K] ----
  {
    dim3 gDD(DMODEL / 32, DMODEL / 32, NLAYER);
    transpose_bf16_kernel<<<gDD, 256, 0, stream>>>(Wq, WqT, DMODEL, DMODEL);
    transpose_bf16_kernel<<<gDD, 256, 0, stream>>>(Wk, WkT, DMODEL, DMODEL);
    transpose_bf16_kernel<<<gDD, 256, 0, stream>>>(Wv, WvT, DMODEL, DMODEL);
    transpose_bf16_kernel<<<gDD, 256, 0, stream>>>(Wo, WoT, DMODEL, DMODEL);
    dim3 gDF(DMODEL / 32, FFDIM / 32, NLAYER);
    transpose_bf16_kernel<<<gDF, 256, 0, stream>>>(W1, W1T, DMODEL, FFDIM);
    dim3 gFD(FFDIM / 32, DMODEL / 32, NLAYER);
    transpose_bf16_kernel<<<gFD, 256, 0, stream>>>(W2, W2T, FFDIM, DMODEL);
    dim3 gHV(DMODEL / 32, VOCAB / 32, 1);
    transpose_bf16_kernel<<<gHV, 256, 0, stream>>>(head_W, headT, DMODEL, VOCAB);
  }

  // ---- 1) embedding + prompt concat ----
  {
    size_t total = (size_t)M * DMODEL;
    embed_concat_kernel<<<(int)((total + 255) / 256), 256, 0, stream>>>(
        input_ids, embed, prompt, x);
  }

  // ---- 2) transformer layers ----
  for (int l = 0; l < NLAYER; ++l) {
    const __bf16* WqTl = WqT + (size_t)l * DMODEL * DMODEL;
    const __bf16* WkTl = WkT + (size_t)l * DMODEL * DMODEL;
    const __bf16* WvTl = WvT + (size_t)l * DMODEL * DMODEL;
    const __bf16* WoTl = WoT + (size_t)l * DMODEL * DMODEL;
    const __bf16* W1Tl = W1T + (size_t)l * DMODEL * FFDIM;
    const __bf16* W2Tl = W2T + (size_t)l * FFDIM * DMODEL;

    layernorm_kernel<<<M, 256, 0, stream>>>(x, ln1_g + (size_t)l * DMODEL,
                                            ln1_b + (size_t)l * DMODEL, h);

    launch_gemm(h, WqTl, q, bq + (size_t)l * DMODEL, nullptr, M, DMODEL, DMODEL, 0, stream);
    launch_gemm(h, WkTl, k, bk + (size_t)l * DMODEL, nullptr, M, DMODEL, DMODEL, 0, stream);
    launch_gemm(h, WvTl, v, bv + (size_t)l * DMODEL, nullptr, M, DMODEL, DMODEL, 0, stream);

    {
      dim3 grid(TTOT, NHEAD, BATCH);
      attention_kernel<<<grid, 128, 0, stream>>>(q, k, v, attn_mask, o);
    }

    // x = x + o @ Wo + bo
    launch_gemm(o, WoTl, x, bo + (size_t)l * DMODEL, x, M, DMODEL, DMODEL, 0, stream);

    layernorm_kernel<<<M, 256, 0, stream>>>(x, ln2_g + (size_t)l * DMODEL,
                                            ln2_b + (size_t)l * DMODEL, h);

    // ffh = relu(h @ W1 + b1)  (bf16 out)
    launch_gemm(h, W1Tl, ffh, b1 + (size_t)l * FFDIM, nullptr,
                M, FFDIM, DMODEL, GEMM_RELU | GEMM_OUT_BF16, stream);
    // x = x + ffh @ W2 + b2
    launch_gemm(ffh, W2Tl, x, b2 + (size_t)l * DMODEL, x, M, DMODEL, FFDIM, 0, stream);
  }

  // ---- 3) final layernorm + LM head -> logits ----
  layernorm_kernel<<<M, 256, 0, stream>>>(x, lnf_g, lnf_b, h);
  launch_gemm(h, headT, logits, nullptr, nullptr, M, VOCAB, DMODEL, 0, stream);

  // ---- 4) loss ----
  loss_row_kernel<<<BATCH * TLEN, 256, 0, stream>>>(logits, labels, rowloss);
  loss_final_kernel<<<1, 256, 0, stream>>>(rowloss, lossp);
}
